// BinaryDense_6949257085569
// MI455X (gfx1250) — matrix operand order repaired and checked by reading, not compiled, not run on hardware
//
#include <hip/hip_runtime.h>

// ---------------------------------------------------------------------------
// BinaryDense: out = nmk * (x @ f(W)) + bias,  f(W)=tanh(kk*W) or sign(W)
// M=8192, N=4096, K=4096.  bf16x3 split GEMM on v_wmma_f32_16x16x32_bf16,
// tiles staged by the Tensor Data Mover (tensor_load_to_lds + TENSORcnt),
// double-buffered LDS.
// ---------------------------------------------------------------------------

typedef __attribute__((ext_vector_type(16))) __bf16         v16bf;
typedef __attribute__((ext_vector_type(8)))  __bf16         v8bf;
typedef __attribute__((ext_vector_type(8)))  float          v8f;
typedef __attribute__((ext_vector_type(4)))  unsigned int   u32x4;
typedef __attribute__((ext_vector_type(4)))  float          f32x4;
typedef __attribute__((ext_vector_type(4)))  unsigned short u16x4;
typedef __attribute__((ext_vector_type(4)))  unsigned int   tdm_g0_t;
typedef __attribute__((ext_vector_type(8)))  int            tdm_g1_t;
typedef __attribute__((ext_vector_type(4)))  int            tdm_g2_t;

#define M_DIM 8192
#define N_DIM 4096
#define K_DIM 4096
#define NKT   (K_DIM / 32)

#if __has_builtin(__builtin_amdgcn_tensor_load_to_lds)
#define USE_TDM 1
#else
#define USE_TDM 0
#endif

__device__ __forceinline__ unsigned short f2bf(float f) {
  unsigned int u = __float_as_uint(f);
  u += 0x7fffu + ((u >> 16) & 1u);        // round-to-nearest-even
  return (unsigned short)(u >> 16);
}
__device__ __forceinline__ float bf2f(unsigned short h) {
  return __uint_as_float(((unsigned int)h) << 16);
}

// Load a 16-half WMMA fragment from two 16-byte LDS chunks.
__device__ __forceinline__ v16bf ldfrag(const unsigned short* p0,
                                        const unsigned short* p1) {
  struct P { v8bf a, b; } f;
  f.a = *(const v8bf*)p0;
  f.b = *(const v8bf*)p1;
  return __builtin_bit_cast(v16bf, f);
}

#if USE_TDM
// Issue one TDM descriptor: 128-row x 32-element (bf16) 2D tile, global
// row stride K_DIM elements, packed contiguously into LDS (64 B per row).
// D# layout per CDNA5 ISA §8 (group0: count/lds_addr/global_addr/type=2;
// group1: data_size=2B, tile dims, dim0 stride; groups 2/3 zero for 2D).
__device__ __forceinline__ void tdm_load_tile(unsigned lds_byte_addr,
                                              const unsigned short* gptr) {
  unsigned long long ga = (unsigned long long)gptr;
  tdm_g0_t g0 = { 1u,                                   // count=1, no gather
                  lds_byte_addr,                        // LDS dest (bytes)
                  (unsigned)ga,                         // global_addr[31:0]
                  ((unsigned)(ga >> 32) & 0x01FFFFFFu)  // global_addr[56:32]
                      | 0x80000000u };                  // type=2 ("image")
  tdm_g1_t g1 = { (int)(1u << 16),          // data_size=1 (2 bytes); mask=0
                  (int)0xFFFF0000u,         // tensor_dim0[15:0]=0xFFFF (huge)
                  (int)0xFFFF7FFFu,         // tensor_dim0[31:16]|tensor_dim1[15:0]
                  (int)(0x7FFFu | (32u << 16)),  // tensor_dim1[31:16] | tile_dim0=32
                  (int)128,                 // tile_dim1=128, tile_dim2=0
                  (int)K_DIM,               // tensor_dim0_stride[31:0]=4096
                  0,                        // stride0[47:32]=0, stride1[15:0]=0
                  0 };                      // tensor_dim1_stride hi (unused, 2D)
  tdm_g2_t gz = {};
#if __has_include(<hip/amd_detail/amd_gfx1250_TDM.h>)
  tdm_g1_t z8 = {};
  __builtin_amdgcn_tensor_load_to_lds(g0, g1, gz, gz, z8, 0);   // clang-23 form
#else
  __builtin_amdgcn_tensor_load_to_lds(g0, g1, gz, gz, 0);       // ROCm 7.2 form
#endif
}
__device__ __forceinline__ unsigned lds_off(const void* p) {
  return (unsigned)(unsigned long long)p;   // flat LDS addr: low 32 bits = offset
}
#endif

// ---- split x (f32, M x K row-major) into bf16 hi/lo --------------------------
__global__ void prep_x_kernel(const float* __restrict__ x,
                              unsigned short* __restrict__ xh,
                              unsigned short* __restrict__ xl) {
  size_t i = ((size_t)blockIdx.x * blockDim.x + threadIdx.x) * 4;
  f32x4 v = *(const f32x4*)(x + i);
  unsigned short h[4], l[4];
#pragma unroll
  for (int t = 0; t < 4; ++t) {
    h[t] = f2bf(v[t]);
    l[t] = f2bf(v[t] - bf2f(h[t]));
  }
  *(u16x4*)(xh + i) = (u16x4){h[0], h[1], h[2], h[3]};
  *(u16x4*)(xl + i) = (u16x4){l[0], l[1], l[2], l[3]};
}

// ---- f(W), bf16 split, and transpose to N-major (Wt[n][k]) -------------------
__global__ void prep_w_kernel(const float* __restrict__ w,
                              const float* __restrict__ kk_p,
                              unsigned short* __restrict__ wth,
                              unsigned short* __restrict__ wtl) {
  __shared__ unsigned short sh[32][33];   // padded: no bank conflicts
  __shared__ unsigned short sl[32][33];
  const float kk = kk_p[0];
  const bool soft = (kk < 1000.0f);       // uniform branch across dispatch
  const int n0 = blockIdx.x * 32, k0 = blockIdx.y * 32;
  const int tx = threadIdx.x, ty = threadIdx.y;
#pragma unroll
  for (int r = 0; r < 4; ++r) {
    int kl = ty + r * 8;
    float v = w[(size_t)(k0 + kl) * N_DIM + (n0 + tx)];
    float t = soft ? tanhf(v * kk)
                   : (v > 0.f ? 1.f : (v < 0.f ? -1.f : 0.f));
    unsigned short h = f2bf(t);
    sh[kl][tx] = h;
    sl[kl][tx] = f2bf(t - bf2f(h));
  }
  __syncthreads();
#pragma unroll
  for (int r = 0; r < 4; ++r) {
    int nl = ty + r * 8;
    size_t o = (size_t)(n0 + nl) * K_DIM + (k0 + tx);
    wth[o] = sh[tx][nl];
    wtl[o] = sl[tx][nl];
  }
}

// ---- main GEMM: 128x128 block tile, BK=32, 8 waves, bf16x3 WMMA --------------
__global__ void __launch_bounds__(256)
binary_dense_gemm(const unsigned short* __restrict__ xh,
                  const unsigned short* __restrict__ xl,
                  const unsigned short* __restrict__ wth,
                  const unsigned short* __restrict__ wtl,
                  const float* __restrict__ bias,
                  const float* __restrict__ nmk_p,
                  float* __restrict__ out) {
  __shared__ unsigned short sAh[2][128 * 32];
  __shared__ unsigned short sAl[2][128 * 32];
  __shared__ unsigned short sBh[2][128 * 32];   // stored N-major: sB[n][k]
  __shared__ unsigned short sBl[2][128 * 32];

  const int tid   = threadIdx.x;
  const int lane  = tid & 31;
  const int wid   = tid >> 5;       // 0..7
  const int wm    = wid & 3;        // 4 wave-rows of 32 M each
  const int wn    = wid >> 2;       // 2 wave-cols of 64 N each
  const int nlane = lane & 15;
  const int khalf = lane >> 4;      // 0|1: selects lane-half K ranges

  const int m0 = blockIdx.y * 128;
  const int n0 = blockIdx.x * 128;

  const v8f vzero = {};
  v8f acc[2][4];
#pragma unroll
  for (int i = 0; i < 2; ++i)
#pragma unroll
    for (int j = 0; j < 4; ++j) acc[i][j] = vzero;

#if USE_TDM
  // Prologue: wave 0 kicks off the k-tile 0 DMAs into buffer 0.
  if (wid == 0) {
    tdm_load_tile(lds_off(&sAh[0][0]), &xh[(size_t)m0 * K_DIM]);
    tdm_load_tile(lds_off(&sAl[0][0]), &xl[(size_t)m0 * K_DIM]);
    tdm_load_tile(lds_off(&sBh[0][0]), &wth[(size_t)n0 * K_DIM]);
    tdm_load_tile(lds_off(&sBl[0][0]), &wtl[(size_t)n0 * K_DIM]);
  }
#endif

  for (int kt = 0; kt < NKT; ++kt) {
#if USE_TDM
    const int cur = kt & 1;
    // Publish buffer[cur]: wave 0 drains TENSORcnt, then one WG barrier.
    if (wid == 0) __builtin_amdgcn_s_wait_tensorcnt(0);
    __syncthreads();
    // Overlap: issue next k-tile's DMAs into the other buffer while all
    // 8 waves compute on buffer[cur].  (Barrier above guarantees nobody is
    // still reading buffer[cur^1] from the previous iteration.)
    if (wid == 0 && kt + 1 < NKT) {
      const int nxt = cur ^ 1;
      const size_t ka = (size_t)m0 * K_DIM + (kt + 1) * 32;
      const size_t kb = (size_t)n0 * K_DIM + (kt + 1) * 32;
      tdm_load_tile(lds_off(&sAh[nxt][0]), &xh[ka]);
      tdm_load_tile(lds_off(&sAl[nxt][0]), &xl[ka]);
      tdm_load_tile(lds_off(&sBh[nxt][0]), &wth[kb]);
      tdm_load_tile(lds_off(&sBl[nxt][0]), &wtl[kb]);
    }
#else
    const int cur = 0;
    const int k0 = kt * 32;
    // Fallback: per-thread b128 copies (512 16-byte chunks per tile).
#pragma unroll
    for (int cc = 0; cc < 2; ++cc) {
      const int c    = tid + cc * 256;
      const int row  = c >> 2;
      const int koff = (c & 3) * 8;
      const size_t ga = (size_t)(m0 + row) * K_DIM + k0 + koff;
      const size_t gb = (size_t)(n0 + row) * K_DIM + k0 + koff;
      const int sa = row * 32 + koff;
      *(u32x4*)&sAh[0][sa] = *(const u32x4*)&xh[ga];
      *(u32x4*)&sAl[0][sa] = *(const u32x4*)&xl[ga];
      *(u32x4*)&sBh[0][sa] = *(const u32x4*)&wth[gb];
      *(u32x4*)&sBl[0][sa] = *(const u32x4*)&wtl[gb];
    }
    __syncthreads();
#endif

    // --- A fragments (ISA 16-bit A layout: lane-half picks K 0..7/16..23 vs
    //     8..15/24..31; two 16B chunks 16 halves apart) -----------------------
    v16bf Ah[2], Al[2];
#pragma unroll
    for (int i = 0; i < 2; ++i) {
      const int row = wm * 32 + i * 16 + nlane;
      const int kb  = khalf * 8;
      const unsigned short* p = &sAh[cur][row * 32 + kb];
      Ah[i] = ldfrag(p, p + 16);
      const unsigned short* q = &sAl[cur][row * 32 + kb];
      Al[i] = ldfrag(q, q + 16);
    }

    // --- B fragments (same K order as A: lane-half picks K 0..7/16..23 vs
    //     8..15/24..31) + 3 WMMAs per tile ------------------------------------
#pragma unroll
    for (int j = 0; j < 4; ++j) {
      const int col = wn * 64 + j * 16 + nlane;
      const int kb  = khalf * 8;
      const unsigned short* p = &sBh[cur][col * 32 + kb];
      const v16bf Bh = ldfrag(p, p + 16);
      const unsigned short* q = &sBl[cur][col * 32 + kb];
      const v16bf Bl = ldfrag(q, q + 16);
#pragma unroll
      for (int i = 0; i < 2; ++i) {
        acc[i][j] = __builtin_amdgcn_wmma_f32_16x16x32_bf16(
            false, Ah[i], false, Bh, (short)0, acc[i][j], false, false);
        acc[i][j] = __builtin_amdgcn_wmma_f32_16x16x32_bf16(
            false, Ah[i], false, Bl, (short)0, acc[i][j], false, false);
        acc[i][j] = __builtin_amdgcn_wmma_f32_16x16x32_bf16(
            false, Al[i], false, Bh, (short)0, acc[i][j], false, false);
      }
    }
#if !USE_TDM
    __syncthreads();
#endif
  }

  // --- epilogue: D layout VGPR r -> M = r + 8*khalf, N = nlane ---------------
  const float nmk = nmk_p[0];
#pragma unroll
  for (int i = 0; i < 2; ++i) {
#pragma unroll
    for (int j = 0; j < 4; ++j) {
      const int mb = m0 + wm * 32 + i * 16 + khalf * 8;
      const int nn = n0 + wn * 64 + j * 16 + nlane;
      const float bv = bias[nn];
#pragma unroll
      for (int r = 0; r < 8; ++r)
        out[(size_t)(mb + r) * N_DIM + nn] = nmk * acc[i][j][r] + bv;
    }
  }
}

// ---------------------------------------------------------------------------
extern "C" void kernel_launch(void* const* d_in, const int* in_sizes, int n_in,
                              void* d_out, int out_size, void* d_ws, size_t ws_size,
                              hipStream_t stream) {
  const float* x    = (const float*)d_in[0];
  const float* kern = (const float*)d_in[1];
  const float* bias = (const float*)d_in[2];
  const float* nmk  = (const float*)d_in[3];
  const float* kk   = (const float*)d_in[4];
  float* out = (float*)d_out;

  // workspace: xh, xl (M*K bf16 each) + wth, wtl (N*K bf16 each) = 192 MB
  unsigned short* xh  = (unsigned short*)d_ws;
  unsigned short* xl  = xh  + (size_t)M_DIM * K_DIM;
  unsigned short* wth = xl  + (size_t)M_DIM * K_DIM;
  unsigned short* wtl = wth + (size_t)N_DIM * K_DIM;

  const unsigned px_blocks = (unsigned)(((size_t)M_DIM * K_DIM) / 4 / 256);
  prep_x_kernel<<<px_blocks, 256, 0, stream>>>(x, xh, xl);
  prep_w_kernel<<<dim3(N_DIM / 32, K_DIM / 32), dim3(32, 8), 0, stream>>>(
      kern, kk, wth, wtl);
  binary_dense_gemm<<<dim3(N_DIM / 128, M_DIM / 128), 256, 0, stream>>>(
      xh, xl, wth, wtl, bias, nmk, out);
}
